// ProposedModel_89137751262019
// MI455X (gfx1250) — matrix operand order repaired and checked by reading, not compiled, not run on hardware
//
#include <hip/hip_runtime.h>
#include <hip/hip_bf16.h>

typedef __attribute__((ext_vector_type(16))) _Float16 v16h;
typedef __attribute__((ext_vector_type(8)))  _Float16 v8h;
typedef __attribute__((ext_vector_type(8)))  float    v8f;

#define D_     256
#define D3_    768
#define T_     16
#define P_     512
#define BP_    528      // padded patient count: 512 mem + 1 query + 15 pad = 33*16
#define NPAT_  513
#define NQ_    15       // T-1 prefix encoders
#define NMEM_  7680     // 512*15 memory rows
#define V0_    2000
#define V1_    1500
#define V2_    500
#define LD_    24
#define LP_    16
#define LM_    24
#define THRESH_ 0.8f
#define EPS_    1e-6f
#define KSEL_   16

// ---------------------------------------------------------------------------
// WMMA fragment helpers (16x16x32 f16; contiguous k order used here, same in A and B)
// A 16x32 (MxK): lane m=lane&15; halves 0..15 = K[k0+16*hi .. +15] of row m
// B 32x16 (KxN): lane n=lane&15; halves 0..15 = K[k0+16*hi .. +15] of weight row n
// C/D f32 16x16: element r -> row r+8*hi, col lane&15
// ---------------------------------------------------------------------------
__device__ __forceinline__ v16h load_a_frag(const _Float16* base, int row_stride,
                                            int k0, int lane) {
  int m = lane & 15, hi = (lane >> 4) & 1;
  return *(const v16h*)(base + (size_t)m * row_stride + k0 + hi * 16);
}

__device__ __forceinline__ v16h load_b_frag(const _Float16* Wbase, int row_stride,
                                            int k0, int lane) {
  int n = lane & 15, hi = (lane >> 4) & 1;
  return *(const v16h*)(Wbase + (size_t)n * row_stride + k0 + hi * 16);
}

__device__ __forceinline__ float sigm(float x) { return 1.f / (1.f + __expf(-x)); }

// ---------------------------------------------------------------------------
// 1) Embedding bag-sums: e_nomed (f32) and e_full (f16) per (patient, visit)
// ---------------------------------------------------------------------------
__global__ void embed_kernel(const int* qd, const int* qp, const int* qm,
                             const int* md, const int* mp, const int* mm,
                             const float* emb,
                             _Float16* e_full_h, float* e_nomed) {
  int bid = blockIdx.x;           // p*16 + t, p in [0, BP_)
  int p = bid >> 4, t = bid & 15;
  int d = threadIdx.x;
  float sdp = 0.f, sm = 0.f;
  if (p < NPAT_) {
    const int *diag, *proc, *med;
    if (p < P_) {
      diag = md + ((size_t)p * T_ + t) * LD_;
      proc = mp + ((size_t)p * T_ + t) * LP_;
      med  = mm + ((size_t)p * T_ + t) * LM_;
    } else {
      diag = qd + (size_t)t * LD_;
      proc = qp + (size_t)t * LP_;
      med  = qm + (size_t)t * LM_;
    }
    for (int i = 0; i < LD_; ++i) sdp += emb[(size_t)diag[i] * D_ + d];
    for (int i = 0; i < LP_; ++i) sdp += emb[((size_t)proc[i] + V0_) * D_ + d];
    for (int i = 0; i < LM_; ++i) sm  += emb[((size_t)med[i] + V0_ + V1_) * D_ + d];
  }
  size_t o = (size_t)bid * D_ + d;
  e_nomed[o]  = sdp;
  e_full_h[o] = (_Float16)(sdp + sm);
}

// ---------------------------------------------------------------------------
// 2) f32 -> f16 weight conversion
// ---------------------------------------------------------------------------
__global__ void cvt_f16_kernel(const float* src, _Float16* dst, int n) {
  int i = blockIdx.x * blockDim.x + threadIdx.x;
  if (i < n) dst[i] = (_Float16)src[i];
}

// ---------------------------------------------------------------------------
// 3) gx = e_full @ Wih^T : [8448 x 768], K=256.  One wave per 16x16 tile.
// ---------------------------------------------------------------------------
__global__ void gx_gemm_kernel(const _Float16* efh, const _Float16* wih_h, float* gx) {
  int gw = blockIdx.x * 8 + (threadIdx.x >> 5);     // tile id over 528*48
  int lane = threadIdx.x & 31;
  int tm = gw / 48, tn = gw % 48;
  const _Float16* A = efh   + (size_t)tm * 16 * D_;
  const _Float16* B = wih_h + (size_t)tn * 16 * D_;
  v8f acc = {};
#pragma unroll
  for (int k = 0; k < D_; k += 32) {
    v16h a = load_a_frag(A, D_, k, lane);
    v16h b = load_b_frag(B, D_, k, lane);
    acc = __builtin_amdgcn_wmma_f32_16x16x32_f16(false, a, false, b, (short)0, acc,
                                                 false, false);
  }
  int n = lane & 15, hi = lane >> 4;
#pragma unroll
  for (int r = 0; r < 8; ++r) {
    size_t row = (size_t)tm * 16 + r + 8 * hi;
    gx[row * D3_ + tn * 16 + n] = acc[r];
  }
}

// ---------------------------------------------------------------------------
// 4) One GRU step: gh = h_{t-1} @ Whh^T into LDS (WMMA), then gate math.
//    Block = 16 patients; 8 waves x 6 column tiles cover the 768 gate dims.
// ---------------------------------------------------------------------------
__global__ void gru_step_kernel(const _Float16* h_h, const float* h_f32,
                                const float* gx, const _Float16* whh_h,
                                const float* bih, const float* bhh,
                                float* h_f32_out, _Float16* h_h_out, int t) {
  __shared__ float gh[16 * D3_];
  int wid = threadIdx.x >> 5, lane = threadIdx.x & 31;
  int p0 = blockIdx.x * 16;
  if (t > 0) {
    const _Float16* A = h_h + ((size_t)p0 * T_ + (t - 1)) * D_;   // stride T_*D_
    for (int j = 0; j < 6; ++j) {
      int tn = wid * 6 + j;
      const _Float16* B = whh_h + (size_t)tn * 16 * D_;
      v8f acc = {};
#pragma unroll
      for (int k = 0; k < D_; k += 32) {
        v16h a = load_a_frag(A, T_ * D_, k, lane);
        v16h b = load_b_frag(B, D_, k, lane);
        acc = __builtin_amdgcn_wmma_f32_16x16x32_f16(false, a, false, b, (short)0, acc,
                                                     false, false);
      }
      int n = lane & 15, hi = lane >> 4;
#pragma unroll
      for (int r = 0; r < 8; ++r)
        gh[(r + 8 * hi) * D3_ + tn * 16 + n] = acc[r];
    }
  } else {
    for (int i = threadIdx.x; i < 16 * D3_; i += 256) gh[i] = 0.f;
  }
  __syncthreads();

  int d = threadIdx.x;
  for (int m = 0; m < 16; ++m) {
    size_t row = (size_t)p0 + m;
    size_t go = (row * T_ + t) * D3_;
    float xr = gx[go + d]       + bih[d];
    float xz = gx[go + 256 + d] + bih[256 + d];
    float xn = gx[go + 512 + d] + bih[512 + d];
    float hr = gh[m * D3_ + d]       + bhh[d];
    float hz = gh[m * D3_ + 256 + d] + bhh[256 + d];
    float hn = gh[m * D3_ + 512 + d] + bhh[512 + d];
    float r  = sigm(xr + hr);
    float z  = sigm(xz + hz);
    float ng = tanhf(xn + r * hn);
    float hold = (t > 0) ? h_f32[(row * T_ + (t - 1)) * D_ + d] : 0.f;
    float hv = (1.f - z) * ng + z * hold;
    size_t ho = (row * T_ + t) * D_ + d;
    h_f32_out[ho] = hv;
    h_h_out[ho]   = (_Float16)hv;
  }
}

// ---------------------------------------------------------------------------
// 5) terms = tanh(h @ W2^T + b2) * e_full   (WMMA GEMM with fused epilogue)
// ---------------------------------------------------------------------------
__global__ void w2_terms_kernel(const _Float16* h_h, const _Float16* w2_h,
                                const float* b2, const _Float16* efh, float* terms) {
  int gw = blockIdx.x * 8 + (threadIdx.x >> 5);   // tile id over 528*16
  int lane = threadIdx.x & 31;
  int tm = gw >> 4, tn = gw & 15;
  const _Float16* A = h_h  + (size_t)tm * 16 * D_;
  const _Float16* B = w2_h + (size_t)tn * 16 * D_;
  v8f acc = {};
#pragma unroll
  for (int k = 0; k < D_; k += 32) {
    v16h a = load_a_frag(A, D_, k, lane);
    v16h b = load_b_frag(B, D_, k, lane);
    acc = __builtin_amdgcn_wmma_f32_16x16x32_f16(false, a, false, b, (short)0, acc,
                                                 false, false);
  }
  int n = lane & 15, hi = lane >> 4;
  int col = tn * 16 + n;
  float bb = b2[col];
#pragma unroll
  for (int r = 0; r < 8; ++r) {
    size_t row = (size_t)tm * 16 + r + 8 * hi;
    float ef = (float)efh[row * D_ + col];
    terms[row * D_ + col] = tanhf(acc[r] + bb) * ef;
  }
}

// ---------------------------------------------------------------------------
// 6) Prefix cumsum over visits + enc assembly + row norms + normalized f16 rows
// ---------------------------------------------------------------------------
__global__ void cumsum_enc_kernel(const float* terms, const float* e_nomed,
                                  float* ENC, _Float16* Mn_h, _Float16* Qn_h) {
  __shared__ float red[256];
  int p = blockIdx.x;             // 0..512
  int d = threadIdx.x;
  float c = 0.f;
  for (int i = 0; i < NQ_; ++i) {
    c += terms[((size_t)p * T_ + i) * D_ + d];
    float e = c + e_nomed[((size_t)p * T_ + i + 1) * D_ + d];
    size_t erow = (size_t)p * NQ_ + i;
    ENC[erow * D_ + d] = e;
    red[d] = e * e;
    __syncthreads();
    for (int s = 128; s > 0; s >>= 1) {
      if (d < s) red[d] += red[d + s];
      __syncthreads();
    }
    float nrm = fmaxf(sqrtf(red[0]), EPS_);
    __syncthreads();
    float en = e / nrm;
    if (p < P_) Mn_h[erow * D_ + d]     = (_Float16)en;
    else        Qn_h[(size_t)i * D_ + d] = (_Float16)en;
  }
  if (p == 0) Qn_h[(size_t)NQ_ * D_ + d] = (_Float16)0.f;   // pad query row 15
}

// ---------------------------------------------------------------------------
// 7) Cosine sims = Qn (16x256) @ Mn^T (256x7680)  via WMMA, one wave per tile
// ---------------------------------------------------------------------------
__global__ void sims_kernel(const _Float16* Qn_h, const _Float16* Mn_h, float* sims) {
  int gw = blockIdx.x * 8 + (threadIdx.x >> 5);   // 0..479 column tiles
  int lane = threadIdx.x & 31;
  const _Float16* B = Mn_h + (size_t)gw * 16 * D_;
  v8f acc = {};
#pragma unroll
  for (int k = 0; k < D_; k += 32) {
    v16h a = load_a_frag(Qn_h, D_, k, lane);
    v16h b = load_b_frag(B, D_, k, lane);
    acc = __builtin_amdgcn_wmma_f32_16x16x32_f16(false, a, false, b, (short)0, acc,
                                                 false, false);
  }
  int n = lane & 15, hi = lane >> 4;
#pragma unroll
  for (int r = 0; r < 8; ++r) {
    int q = r + 8 * hi;
    sims[(size_t)q * NMEM_ + gw * 16 + n] = acc[r];
  }
}

// ---------------------------------------------------------------------------
// 8) Top-16 + threshold + weighted gather-sum.  One block per query row.
// ---------------------------------------------------------------------------
__global__ void topk_kernel(const float* sims, const float* ENC, _Float16* R_h) {
  __shared__ float sv[NMEM_];
  __shared__ float rv[256];
  __shared__ int   ri[256];
  int q = blockIdx.x, tid = threadIdx.x;
  for (int i = tid; i < NMEM_; i += 256) sv[i] = sims[(size_t)q * NMEM_ + i];
  float out = ENC[((size_t)P_ * NQ_ + q) * D_ + tid];   // Q row q, dim tid
  __syncthreads();
  for (int it = 0; it < KSEL_; ++it) {
    float bv = -1e30f; int bi = 0;
    for (int i = tid; i < NMEM_; i += 256) {
      float v = sv[i];
      if (v > bv) { bv = v; bi = i; }
    }
    rv[tid] = bv; ri[tid] = bi;
    __syncthreads();
    for (int s = 128; s > 0; s >>= 1) {
      if (tid < s) {
        if (rv[tid + s] > rv[tid] ||
            (rv[tid + s] == rv[tid] && ri[tid + s] < ri[tid])) {
          rv[tid] = rv[tid + s]; ri[tid] = ri[tid + s];
        }
      }
      __syncthreads();
    }
    float w = rv[0]; int sel = ri[0];
    __syncthreads();
    if (w > THRESH_) out += w * ENC[(size_t)sel * D_ + tid];
    if (tid == 0) sv[sel] = -1e30f;
    __syncthreads();
  }
  R_h[(size_t)q * D_ + tid] = (_Float16)out;
  if (q == 0) R_h[(size_t)NQ_ * D_ + tid] = (_Float16)0.f;   // pad row 15
}

// ---------------------------------------------------------------------------
// 9) logits = sigmoid(R @ Wo^T + bo) -> d_out [15 x 500]
// ---------------------------------------------------------------------------
__global__ void logits_kernel(const _Float16* R_h, const _Float16* wo_h,
                              const float* bo, float* out) {
  int gw = blockIdx.x * 8 + (threadIdx.x >> 5);   // 0..31 column tiles
  int lane = threadIdx.x & 31;
  v8f acc = {};
#pragma unroll
  for (int k = 0; k < D_; k += 32) {
    v16h a = load_a_frag(R_h, D_, k, lane);
    int n = lane & 15, hi = (lane >> 4) & 1;
    int rowW = gw * 16 + n;
    if (rowW >= V2_) rowW = V2_ - 1;              // clamp; result discarded at store
    v16h b = *(const v16h*)(wo_h + (size_t)rowW * D_ + k + hi * 16);
    acc = __builtin_amdgcn_wmma_f32_16x16x32_f16(false, a, false, b, (short)0, acc,
                                                 false, false);
  }
  int n = lane & 15, hi = lane >> 4;
  int col = gw * 16 + n;
  if (col < V2_) {
    float bb = bo[col];
#pragma unroll
    for (int r = 0; r < 8; ++r) {
      int q = r + 8 * hi;
      if (q < NQ_) out[(size_t)q * V2_ + col] = sigm(acc[r] + bb);
    }
  }
}

// ---------------------------------------------------------------------------
extern "C" void kernel_launch(void* const* d_in, const int* in_sizes, int n_in,
                              void* d_out, int out_size, void* d_ws, size_t ws_size,
                              hipStream_t stream) {
  const int*   qd   = (const int*)d_in[0];
  const int*   qp   = (const int*)d_in[1];
  const int*   qm   = (const int*)d_in[2];
  const int*   md   = (const int*)d_in[3];
  const int*   mp   = (const int*)d_in[4];
  const int*   mm   = (const int*)d_in[5];
  const float* emb  = (const float*)d_in[6];
  const float* Wih  = (const float*)d_in[7];
  const float* Whh  = (const float*)d_in[8];
  const float* bih  = (const float*)d_in[9];
  const float* bhh  = (const float*)d_in[10];
  const float* W2   = (const float*)d_in[11];
  const float* b2   = (const float*)d_in[12];
  const float* Wo   = (const float*)d_in[13];
  const float* bo   = (const float*)d_in[14];
  float* outp = (float*)d_out;

  char* base = (char*)d_ws;
  size_t off = 0;
  auto carve = [&](size_t bytes) -> void* {
    void* p = base + off;
    off = (off + bytes + 255) & ~(size_t)255;
    return p;
  };
  _Float16* e_full_h = (_Float16*)carve((size_t)BP_ * T_ * D_ * 2);
  float*    e_nomed  = (float*)   carve((size_t)BP_ * T_ * D_ * 4);
  float*    gx       = (float*)   carve((size_t)BP_ * T_ * D3_ * 4);
  float*    h_f32    = (float*)   carve((size_t)BP_ * T_ * D_ * 4);
  _Float16* h_h      = (_Float16*)carve((size_t)BP_ * T_ * D_ * 2);
  float*    terms    = (float*)   carve((size_t)BP_ * T_ * D_ * 4);
  float*    ENC      = (float*)   carve((size_t)NPAT_ * NQ_ * D_ * 4);
  _Float16* Mn_h     = (_Float16*)carve((size_t)NMEM_ * D_ * 2);
  _Float16* Qn_h     = (_Float16*)carve((size_t)16 * D_ * 2);
  float*    sims     = (float*)   carve((size_t)16 * NMEM_ * 4);
  _Float16* R_h      = (_Float16*)carve((size_t)16 * D_ * 2);
  _Float16* wih_h    = (_Float16*)carve((size_t)D3_ * D_ * 2);
  _Float16* whh_h    = (_Float16*)carve((size_t)D3_ * D_ * 2);
  _Float16* w2_h     = (_Float16*)carve((size_t)D_ * D_ * 2);
  _Float16* wo_h     = (_Float16*)carve((size_t)V2_ * D_ * 2);

  // 1) embeddings (also zeroes padded patients 513..527)
  embed_kernel<<<BP_ * T_, 256, 0, stream>>>(qd, qp, qm, md, mp, mm, emb,
                                             e_full_h, e_nomed);
  // 2) weight conversions
  cvt_f16_kernel<<<(D3_ * D_ + 255) / 256, 256, 0, stream>>>(Wih, wih_h, D3_ * D_);
  cvt_f16_kernel<<<(D3_ * D_ + 255) / 256, 256, 0, stream>>>(Whh, whh_h, D3_ * D_);
  cvt_f16_kernel<<<(D_ * D_ + 255) / 256, 256, 0, stream>>>(W2, w2_h, D_ * D_);
  cvt_f16_kernel<<<(V2_ * D_ + 255) / 256, 256, 0, stream>>>(Wo, wo_h, V2_ * D_);
  // 3) gx = X @ Wih^T  : 528 row-tiles * 48 col-tiles / 8 waves per block
  gx_gemm_kernel<<<(33 * 16 * 48) / 8, 256, 0, stream>>>(e_full_h, wih_h, gx);
  // 4) 16 sequential GRU steps
  for (int t = 0; t < T_; ++t)
    gru_step_kernel<<<33, 256, 0, stream>>>(h_h, h_f32, gx, whh_h, bih, bhh,
                                            h_f32, h_h, t);
  // 5) terms = tanh(h @ W2^T + b2) * e_full
  w2_terms_kernel<<<(33 * 16 * 16) / 8, 256, 0, stream>>>(h_h, w2_h, b2,
                                                          e_full_h, terms);
  // 6) cumsum + encoder + norms
  cumsum_enc_kernel<<<NPAT_, 256, 0, stream>>>(terms, e_nomed, ENC, Mn_h, Qn_h);
  // 7) cosine sims
  sims_kernel<<<(NMEM_ / 16) / 8, 256, 0, stream>>>(Qn_h, Mn_h, sims);
  // 8) top-k + weighted gather
  topk_kernel<<<NQ_, 256, 0, stream>>>(sims, ENC, R_h);
  // 9) final logits
  logits_kernel<<<(32) / 8, 256, 0, stream>>>(R_h, wo_h, bo, outp);

  (void)in_sizes; (void)n_in; (void)out_size; (void)ws_size;
}
